// LSAModel_24704651887141
// MI455X (gfx1250) — matrix operand order repaired and checked by reading, not compiled, not run on hardware
//
#include <hip/hip_runtime.h>
#include <cstdint>
#include <cstddef>

// ---------------- types ----------------
typedef __bf16 bf16_t;
typedef __attribute__((ext_vector_type(16))) __bf16 v16bf;
typedef __attribute__((ext_vector_type(8)))  __bf16 v8bf;
typedef __attribute__((ext_vector_type(8)))  float  v8f;

#define Bn   512
#define Tn   128
#define Hn   256
#define H4n  1024
#define DINn 64
#define Ln   2

__device__ __forceinline__ float sigmoidf_(float x) { return 1.f / (1.f + expf(-x)); }

// ---------------- utility kernels ----------------
__global__ void k_cvt_bf16(const float* __restrict__ in, bf16_t* __restrict__ out, int n) {
  int i = blockIdx.x * 256 + threadIdx.x;
  if (i < n) out[i] = (bf16_t)in[i];
}
__global__ void k_zero_bf16(bf16_t* p, int n) {
  int i = blockIdx.x * 256 + threadIdx.x;
  if (i < n) p[i] = (bf16_t)0.f;
}
__global__ void k_zero_f32(float* p, int n) {
  int i = blockIdx.x * 256 + threadIdx.x;
  if (i < n) p[i] = 0.f;
}
__global__ void k_copy_bf16(bf16_t* __restrict__ d, const bf16_t* __restrict__ s, int n) {
  int i = blockIdx.x * 256 + threadIdx.x;
  if (i < n) d[i] = s[i];
}
__global__ void k_copy_f32(float* __restrict__ d, const float* __restrict__ s, int n) {
  int i = blockIdx.x * 256 + threadIdx.x;
  if (i < n) d[i] = s[i];
}
__global__ void k_add_f32(const float* __restrict__ a, const float* __restrict__ b,
                          float* __restrict__ o, int n) {
  int i = blockIdx.x * 256 + threadIdx.x;
  if (i < n) o[i] = a[i] + b[i];
}
__global__ void k_pad_f32(const float* __restrict__ b, float* __restrict__ o, int n, int npad) {
  int i = blockIdx.x * 256 + threadIdx.x;
  if (i < npad) o[i] = (i < n) ? b[i] : 0.f;
}

// Pack a row-major (N,K) f32 weight into bf16 WMMA B-operand fragments.
// Fragment (nb,kc): 32 lanes x 16 bf16, lane-contiguous so the GEMM loads
// one 32-byte v16bf per lane per K-chunk.
__global__ void k_pack_b(const float* __restrict__ W, bf16_t* __restrict__ out,
                         int N, int K, int Npad, int Kpad) {
  long long idx = (long long)blockIdx.x * 256 + threadIdx.x;
  long long total = (long long)(Npad / 16) * (Kpad / 32) * 512;
  if (idx >= total) return;
  int e = (int)(idx & 511);
  long long frag = idx >> 9;
  int nK = Kpad >> 5;
  int kc = (int)(frag % nK);
  int nb = (int)(frag / nK);
  int lane = e >> 4, i = e & 15;
  int colv = lane & 15, kb = (lane >> 4) * 8;
  int v = i >> 1, r = i & 1;
  int k = kc * 32 + kb + (i & 7) + ((i >> 3) << 4);
  int n = nb * 16 + colv;
  float val = (n < N && k < K) ? W[(long long)n * K + k] : 0.f;
  out[idx] = (bf16_t)val;
}

// ---------------- WMMA bf16 GEMM ----------------
// C(MxNpad) = act( A(MxKp) @ Bpacked + bias )  [or C += A@B when ACC]
// block = 8 waves, tile = 128 rows x 16 cols; each wave owns one 16x16 tile.
template <int ACT, bool ACC, bool OF32, bool OB16>
__global__ __launch_bounds__(256) void k_gemm(
    const bf16_t* __restrict__ A, int lda,
    const bf16_t* __restrict__ Bp,
    const float* __restrict__ bias,
    float* __restrict__ Cf, bf16_t* __restrict__ Cb,
    int ldc, int M, int Kp) {
  int wave = threadIdx.x >> 5;
  int lane = threadIdx.x & 31;
  int nb = blockIdx.x;
  int rowBase = (blockIdx.y * 8 + wave) * 16;
  if (rowBase >= M) return;
  int col = lane & 15;
  int hi = lane >> 4;
  int nK = Kp >> 5;
  int cidx = nb * 16 + col;
  const bf16_t* bp = Bp + ((size_t)nb * nK) * 512 + lane * 16;
  const bf16_t* arow = A + (size_t)(rowBase + col) * lda;

  v8f acc;
  if (ACC) {
#pragma unroll
    for (int v = 0; v < 8; ++v)
      acc[v] = Cf[(size_t)(rowBase + v + 8 * hi) * ldc + cidx];
  } else {
    float bv = bias ? bias[cidx] : 0.f;
#pragma unroll
    for (int v = 0; v < 8; ++v) acc[v] = bv;
  }

  for (int kc = 0; kc < Kp; kc += 32) {
    const bf16_t* ap = arow + kc + hi * 8;
    v8bf alo = *(const v8bf*)ap;
    v8bf ahi = *(const v8bf*)(ap + 16);
    v16bf af = __builtin_shufflevector(alo, ahi, 0, 1, 2, 3, 4, 5, 6, 7,
                                       8, 9, 10, 11, 12, 13, 14, 15);
    v16bf bfrag = *(const v16bf*)bp;
    __builtin_prefetch(bp + 512, 0, 1);  // global_prefetch_b8 for next fragment
    bp += 512;
    acc = __builtin_amdgcn_wmma_f32_16x16x32_bf16(
        false, af, false, bfrag, (short)0, acc, false, false);
  }

#pragma unroll
  for (int v = 0; v < 8; ++v) {
    int r = rowBase + v + 8 * hi;
    float val = acc[v];
    if (ACT == 1) val = fmaxf(val, 0.f);
    if (ACT == 2) val = tanhf(val);
    if (OF32) Cf[(size_t)r * ldc + cidx] = val;
    if (OB16) Cb[(size_t)r * ldc + cidx] = (bf16_t)val;
  }
}

// ---------------- LSTM pointwise ----------------
__global__ void k_lstm(const float* __restrict__ G, float* __restrict__ c,
                       bf16_t* __restrict__ h, bf16_t* __restrict__ hist, int n) {
  int i = blockIdx.x * 256 + threadIdx.x;
  if (i >= n) return;
  int b = i >> 8, d = i & 255;  // Hn == 256
  const float* g = G + (size_t)b * H4n;
  float ig = sigmoidf_(g[d]);
  float fg = sigmoidf_(g[Hn + d]);
  float gg = tanhf(g[2 * Hn + d]);
  float og = sigmoidf_(g[3 * Hn + d]);
  float cn = fg * c[i] + ig * gg;
  float hn = og * tanhf(cn);
  c[i] = cn;
  bf16_t hb = (bf16_t)hn;
  h[i] = hb;
  if (hist) hist[i] = hb;
}

// ---------------- attention ctx (batched GEMV, bandwidth bound) ----------------
// keys/vals laid out (T,B,H) bf16. Writes cat[b] = [ctx | hq].
__global__ __launch_bounds__(256) void k_attn_ctx(
    const bf16_t* __restrict__ hq, const bf16_t* __restrict__ keys,
    const bf16_t* __restrict__ vals, int tcount, bf16_t* __restrict__ cat) {
  __shared__ float sh[Hn];
  __shared__ float sw[Tn];
  int b = blockIdx.x, tid = threadIdx.x;
  sh[tid] = (float)hq[(size_t)b * Hn + tid];
  __syncthreads();
  if (tid < tcount) {
    const bf16_t* kp = keys + (size_t)tid * Bn * Hn + (size_t)b * Hn;
    float s = 0.f;
    for (int d = 0; d < Hn; d += 8) {
      v8bf kv = *(const v8bf*)(kp + d);
#pragma unroll
      for (int j = 0; j < 8; ++j) s += (float)kv[j] * sh[d + j];
    }
    sw[tid] = s;
  }
  __syncthreads();
  if (tid == 0) {
    float m = -1e30f;
    for (int t = 0; t < tcount; ++t) m = fmaxf(m, sw[t]);
    float sum = 0.f;
    for (int t = 0; t < tcount; ++t) { float e = expf(sw[t] - m); sw[t] = e; sum += e; }
    float inv = 1.f / sum;
    for (int t = 0; t < tcount; ++t) sw[t] *= inv;
  }
  __syncthreads();
  float ctx = 0.f;
  for (int t = 0; t < tcount; ++t)
    ctx += sw[t] * (float)vals[(size_t)t * Bn * Hn + (size_t)b * Hn + tid];
  cat[(size_t)b * 2 * Hn + tid] = (bf16_t)ctx;
  cat[(size_t)b * 2 * Hn + Hn + tid] = (bf16_t)sh[tid];
}

// ---------------- final transpose (T,B,16pad) -> (B,T,3) ----------------
__global__ void k_out(const float* __restrict__ L3, float* __restrict__ out, int n) {
  int i = blockIdx.x * 256 + threadIdx.x;
  if (i >= n) return;
  int j = i % 3;
  int bt = i / 3;
  int t = bt % Tn;
  int b = bt / Tn;
  out[i] = L3[((size_t)t * Bn + b) * 16 + j];
}

// ---------------- host ----------------
extern "C" void kernel_launch(void* const* d_in, const int* in_sizes, int n_in,
                              void* d_out, int out_size, void* d_ws, size_t ws_size,
                              hipStream_t stream) {
  (void)in_sizes; (void)n_in; (void)out_size; (void)ws_size;

  const float* x_enc = (const float*)d_in[0];
  const float* x_dec = (const float*)d_in[1];
  const float* enc_Wih[2] = {(const float*)d_in[2],  (const float*)d_in[3]};
  const float* enc_Whh[2] = {(const float*)d_in[4],  (const float*)d_in[5]};
  const float* enc_bih[2] = {(const float*)d_in[6],  (const float*)d_in[7]};
  const float* enc_bhh[2] = {(const float*)d_in[8],  (const float*)d_in[9]};
  const float* enc_Wsa[2] = {(const float*)d_in[10], (const float*)d_in[11]};
  const float* enc_bsa[2] = {(const float*)d_in[12], (const float*)d_in[13]};
  const float* dec_Wih[2] = {(const float*)d_in[14], (const float*)d_in[15]};
  const float* dec_Whh[2] = {(const float*)d_in[16], (const float*)d_in[17]};
  const float* dec_bih[2] = {(const float*)d_in[18], (const float*)d_in[19]};
  const float* dec_bhh[2] = {(const float*)d_in[20], (const float*)d_in[21]};
  const float* dec_Wsa[2] = {(const float*)d_in[22], (const float*)d_in[23]};
  const float* dec_bsa[2] = {(const float*)d_in[24], (const float*)d_in[25]};
  const float* dec_Wia[2] = {(const float*)d_in[26], (const float*)d_in[27]};
  const float* dec_bia[2] = {(const float*)d_in[28], (const float*)d_in[29]};
  const float* headW1 = (const float*)d_in[30];
  const float* headb1 = (const float*)d_in[31];
  const float* headW2 = (const float*)d_in[32];
  const float* headb2 = (const float*)d_in[33];
  const float* headW3 = (const float*)d_in[34];
  const float* headb3 = (const float*)d_in[35];

  // ---- workspace layout ----
  char* base = (char*)d_ws;
  size_t off = 0;
  auto alloc = [&](size_t bytes) -> char* {
    size_t o = (off + 255) & ~(size_t)255;
    off = o + bytes;
    return base + o;
  };
  const size_t BH = (size_t)Bn * Hn;
  const size_t LTBH = (size_t)Ln * Tn * BH;

  bf16_t* xe = (bf16_t*)alloc((size_t)Bn * Tn * DINn * 2);
  bf16_t* xd = (bf16_t*)alloc((size_t)Bn * Tn * DINn * 2);

  bf16_t* pw_eih[2] = {(bf16_t*)alloc(64 * 1024 * 2),  (bf16_t*)alloc(256 * 1024 * 2)};
  bf16_t* pw_ehh[2] = {(bf16_t*)alloc(256 * 1024 * 2), (bf16_t*)alloc(256 * 1024 * 2)};
  bf16_t* pw_esa[2] = {(bf16_t*)alloc(512 * 256 * 2),  (bf16_t*)alloc(512 * 256 * 2)};
  bf16_t* pw_dih[2] = {(bf16_t*)alloc(64 * 1024 * 2),  (bf16_t*)alloc(256 * 1024 * 2)};
  bf16_t* pw_dhh[2] = {(bf16_t*)alloc(256 * 1024 * 2), (bf16_t*)alloc(256 * 1024 * 2)};
  bf16_t* pw_dsa[2] = {(bf16_t*)alloc(512 * 256 * 2),  (bf16_t*)alloc(512 * 256 * 2)};
  bf16_t* pw_dia[2] = {(bf16_t*)alloc(512 * 256 * 2),  (bf16_t*)alloc(512 * 256 * 2)};
  bf16_t* pw_h1 = (bf16_t*)alloc(256 * 224 * 2);
  bf16_t* pw_h2 = (bf16_t*)alloc(224 * 64 * 2);
  bf16_t* pw_h3 = (bf16_t*)alloc(64 * 16 * 2);

  float* blstm_e[2] = {(float*)alloc(1024 * 4), (float*)alloc(1024 * 4)};
  float* blstm_d[2] = {(float*)alloc(1024 * 4), (float*)alloc(1024 * 4)};
  float* hb1 = (float*)alloc(224 * 4);
  float* hb2 = (float*)alloc(64 * 4);
  float* hb3 = (float*)alloc(16 * 4);

  bf16_t* enc_raw = (bf16_t*)alloc(LTBH * 2);
  bf16_t* enc_ref = (bf16_t*)alloc(LTBH * 2);
  bf16_t* dec_pre = (bf16_t*)alloc(LTBH * 2);
  bf16_t* dec_ref = (bf16_t*)alloc(LTBH * 2);

  bf16_t* h_enc = (bf16_t*)alloc(Ln * BH * 2);
  bf16_t* h_dec = (bf16_t*)alloc(Ln * BH * 2);
  float* c_enc = (float*)alloc(Ln * BH * 4);
  float* c_dec = (float*)alloc(Ln * BH * 4);
  float* G = (float*)alloc((size_t)Bn * H4n * 4);
  bf16_t* cat = (bf16_t*)alloc((size_t)Bn * 2 * Hn * 2);

  bf16_t* buf1 = (bf16_t*)alloc((size_t)Tn * Bn * 224 * 2);
  bf16_t* buf2 = (bf16_t*)alloc((size_t)Tn * Bn * 64 * 2);
  float* buf3 = (float*)alloc((size_t)Tn * Bn * 16 * 4);

  // ---- setup: convert inputs, pack weights, build biases, zero state ----
  {
    int n = Bn * Tn * DINn;
    k_cvt_bf16<<<(n + 255) / 256, 256, 0, stream>>>(x_enc, xe, n);
    k_cvt_bf16<<<(n + 255) / 256, 256, 0, stream>>>(x_dec, xd, n);
  }
  auto pack = [&](const float* W, bf16_t* out, int N, int K, int Npad, int Kpad) {
    long long total = (long long)(Npad / 16) * (Kpad / 32) * 512;
    k_pack_b<<<(int)((total + 255) / 256), 256, 0, stream>>>(W, out, N, K, Npad, Kpad);
  };
  for (int l = 0; l < Ln; ++l) {
    int kin = (l == 0) ? DINn : Hn;
    pack(enc_Wih[l], pw_eih[l], H4n, kin, H4n, kin);
    pack(enc_Whh[l], pw_ehh[l], H4n, Hn, H4n, Hn);
    pack(enc_Wsa[l], pw_esa[l], Hn, 2 * Hn, Hn, 2 * Hn);
    pack(dec_Wih[l], pw_dih[l], H4n, kin, H4n, kin);
    pack(dec_Whh[l], pw_dhh[l], H4n, Hn, H4n, Hn);
    pack(dec_Wsa[l], pw_dsa[l], Hn, 2 * Hn, Hn, 2 * Hn);
    pack(dec_Wia[l], pw_dia[l], Hn, 2 * Hn, Hn, 2 * Hn);
    k_add_f32<<<4, 256, 0, stream>>>(enc_bih[l], enc_bhh[l], blstm_e[l], H4n);
    k_add_f32<<<4, 256, 0, stream>>>(dec_bih[l], dec_bhh[l], blstm_d[l], H4n);
  }
  pack(headW1, pw_h1, 200, 256, 224, 256);
  pack(headW2, pw_h2, 50, 200, 64, 224);
  pack(headW3, pw_h3, 3, 50, 16, 64);
  k_pad_f32<<<1, 256, 0, stream>>>(headb1, hb1, 200, 224);
  k_pad_f32<<<1, 256, 0, stream>>>(headb2, hb2, 50, 64);
  k_pad_f32<<<1, 256, 0, stream>>>(headb3, hb3, 3, 16);

  {
    int n = Ln * Bn * Hn;
    k_zero_bf16<<<(n + 255) / 256, 256, 0, stream>>>(h_enc, n);
    k_zero_f32<<<(n + 255) / 256, 256, 0, stream>>>(c_enc, n);
  }

  const int nBH = (Bn * Hn + 255) / 256;
  const dim3 gGate(H4n / 16, Bn / 128);  // 64 x 4
  const dim3 gProj(Hn / 16, Bn / 128);   // 16 x 4

  // ---- encoder ----
  for (int ti = 0; ti < Tn; ++ti) {
    for (int l = 0; l < Ln; ++l) {
      const bf16_t* Ain;
      int lda, Kin;
      const bf16_t* pih;
      if (l == 0) { Ain = xe + (size_t)ti * DINn; lda = Tn * DINn; pih = pw_eih[0]; Kin = DINn; }
      else        { Ain = enc_ref + (size_t)ti * BH; lda = Hn; pih = pw_eih[1]; Kin = Hn; }
      k_gemm<0, false, true, false><<<gGate, 256, 0, stream>>>(
          Ain, lda, pih, blstm_e[l], G, nullptr, H4n, Bn, Kin);
      k_gemm<0, true, true, false><<<gGate, 256, 0, stream>>>(
          h_enc + l * BH, Hn, pw_ehh[l], nullptr, G, nullptr, H4n, Bn, Hn);
      k_lstm<<<nBH, 256, 0, stream>>>(G, c_enc + l * BH, h_enc + l * BH,
                                      enc_raw + ((size_t)l * Tn + ti) * BH, Bn * Hn);
      bf16_t* refslot = enc_ref + ((size_t)l * Tn + ti) * BH;
      if (ti == 0) {
        k_copy_bf16<<<nBH, 256, 0, stream>>>(refslot, h_enc + l * BH, Bn * Hn);
      } else {
        k_attn_ctx<<<Bn, 256, 0, stream>>>(h_enc + l * BH,
                                           enc_raw + (size_t)l * Tn * BH,
                                           enc_ref + (size_t)l * Tn * BH, ti, cat);
        k_gemm<2, false, false, true><<<gProj, 256, 0, stream>>>(
            cat, 2 * Hn, pw_esa[l], enc_bsa[l], nullptr, refslot, Hn, Bn, 2 * Hn);
      }
    }
  }

  // ---- decoder init ----
  {
    int n = Ln * Bn * Hn;
    k_copy_bf16<<<(n + 255) / 256, 256, 0, stream>>>(h_dec, h_enc, n);
    k_copy_f32<<<(n + 255) / 256, 256, 0, stream>>>(c_dec, c_enc, n);
  }
  const bf16_t* enc_out = enc_ref + (size_t)(Ln - 1) * Tn * BH;  // top layer history (T,B,H)

  // ---- decoder ----
  for (int ti = 0; ti < Tn; ++ti) {
    for (int l = 0; l < Ln; ++l) {
      const bf16_t* Ain;
      int lda, Kin;
      const bf16_t* pih;
      if (l == 0) { Ain = xd + (size_t)ti * DINn; lda = Tn * DINn; pih = pw_dih[0]; Kin = DINn; }
      else        { Ain = dec_ref + (size_t)ti * BH; lda = Hn; pih = pw_dih[1]; Kin = Hn; }
      k_gemm<0, false, true, false><<<gGate, 256, 0, stream>>>(
          Ain, lda, pih, blstm_d[l], G, nullptr, H4n, Bn, Kin);
      k_gemm<0, true, true, false><<<gGate, 256, 0, stream>>>(
          h_dec + l * BH, Hn, pw_dhh[l], nullptr, G, nullptr, H4n, Bn, Hn);
      k_lstm<<<nBH, 256, 0, stream>>>(G, c_dec + l * BH, h_dec + l * BH, nullptr, Bn * Hn);

      // inter-attention over full encoder output -> h_ia (stored into pre history)
      bf16_t* hia = dec_pre + ((size_t)l * Tn + ti) * BH;
      k_attn_ctx<<<Bn, 256, 0, stream>>>(h_dec + l * BH, enc_out, enc_out, Tn, cat);
      k_gemm<2, false, false, true><<<gProj, 256, 0, stream>>>(
          cat, 2 * Hn, pw_dia[l], dec_bia[l], nullptr, hia, Hn, Bn, 2 * Hn);

      bf16_t* refslot = dec_ref + ((size_t)l * Tn + ti) * BH;
      if (ti == 0) {
        k_copy_bf16<<<nBH, 256, 0, stream>>>(refslot, hia, Bn * Hn);
      } else {
        k_attn_ctx<<<Bn, 256, 0, stream>>>(hia,
                                           dec_pre + (size_t)l * Tn * BH,
                                           dec_ref + (size_t)l * Tn * BH, ti, cat);
        k_gemm<2, false, false, true><<<gProj, 256, 0, stream>>>(
            cat, 2 * Hn, pw_dsa[l], dec_bsa[l], nullptr, refslot, Hn, Bn, 2 * Hn);
      }
    }
  }

  // ---- head: (T*B,H) -> 224 relu -> 64 relu -> 16 ----
  const bf16_t* top = dec_ref + (size_t)(Ln - 1) * Tn * BH;  // (T*B, H)
  const int Mh = Tn * Bn;
  k_gemm<1, false, false, true><<<dim3(224 / 16, Mh / 128), 256, 0, stream>>>(
      top, Hn, pw_h1, hb1, nullptr, buf1, 224, Mh, 256);
  k_gemm<1, false, false, true><<<dim3(64 / 16, Mh / 128), 256, 0, stream>>>(
      buf1, 224, pw_h2, hb2, nullptr, buf2, 64, Mh, 224);
  k_gemm<0, false, true, false><<<dim3(1, Mh / 128), 256, 0, stream>>>(
      buf2, 64, pw_h3, hb3, buf3, nullptr, 16, Mh, 64);

  // ---- (T,B,16) -> (B,T,3) ----
  int nout = Bn * Tn * 3;
  k_out<<<(nout + 255) / 256, 256, 0, stream>>>(buf3, (float*)d_out, nout);
}
